// ConvLayer_61890478736068
// MI455X (gfx1250) — matrix operand order repaired and checked by reading, not compiled, not run on hardware
//
#include <hip/hip_runtime.h>

// ---------------------------------------------------------------------------
// Implicit-GEMM 3x3 conv (same pad) for MI455X / gfx1250 using
// v_wmma_f32_16x16x32_bf16 (bf16 inputs, f32 accumulate).
//   x : [32][192][56][56] f32
//   W : [384][192][3][3]  f32 (OIHW)
//   b : [384]             f32
// out : [32][384][56][56] f32
//
// GEMM view: M = 32*56*56 pixels, N = 384, K = 192*9.
// Block tile: 128 M (8x8 spatial block of TWO images) x 64 N, 8 waves,
// each wave computes a 32x32 tile (2 M-subtiles x 2 N-subtiles): every
// 4 WMMAs need only 8 ds_load_b128 (2 loads per WMMA, 2x register reuse).
// LDS rows padded to 40 bf16 (80 B) so half-wave fragment reads stride 20
// banks mod 64 -> conflict-free (64-bank, 4B-bank LDS), while keeping the
// 16-B alignment ds_load_b128 requires.
// ---------------------------------------------------------------------------

typedef __bf16 bf16_t;
typedef __attribute__((ext_vector_type(16))) bf16_t v16bf;
typedef __attribute__((ext_vector_type(8)))  float  v8f;

#define C_IN   192
#define H_IN   56
#define W_IN   56
#define C_OUT  384
#define N_IMG  32
#define KPOS   9

#define PCELL  40                 // padded bf16 per patch cell (32 data + 8 pad)
#define BROW   40                 // padded bf16 per btile row  (32 data + 8 pad)
#define PIMG   (100 * PCELL)      // u16 per staged image patch (10x10 cells)
#define BPOS   (64 * BROW)        // u16 per staged weight tap  (64 co rows)

__device__ __forceinline__ unsigned short f2bf(float f) {
  // round-to-nearest-even fp32 -> bf16
  unsigned int u = __float_as_uint(f);
  u += 0x7FFFu + ((u >> 16) & 1u);
  return (unsigned short)(u >> 16);
}

// Wt[pos][c][co] = bf16( W[co][c][kh][kw] ),  pos = kh*3+kw
__global__ void weight_transform(const float* __restrict__ W,
                                 unsigned short* __restrict__ Wt) {
  int idx = blockIdx.x * 256 + threadIdx.x;
  if (idx >= KPOS * C_IN * C_OUT) return;
  int co  = idx % C_OUT;
  int t   = idx / C_OUT;
  int c   = t % C_IN;
  int pos = t / C_IN;
  Wt[idx] = f2bf(W[(co * C_IN + c) * KPOS + pos]);
}

__launch_bounds__(256)
__global__ void conv3x3_wmma_bf16(const float* __restrict__ x,
                                  const unsigned short* __restrict__ Wt,
                                  const float* __restrict__ bias,
                                  float* __restrict__ out) {
  // LDS: two input halo patches [img 2][row 10][col 10][PCELL]
  //      and weights for all 9 taps [pos 9][co 64][BROW]
  __shared__ __align__(16) unsigned short patch[2 * PIMG];      // 16000 B
  __shared__ __align__(16) unsigned short btile[KPOS * BPOS];   // 46080 B

  const int tid  = threadIdx.x;
  const int lane = tid & 31;
  const int wave = tid >> 5;      // 8 waves
  const int wrow = wave >> 1;     // 0..3 : 32-row M group (img = wrow>>1)
  const int wcol = wave & 1;      // 0..1 : 32-col N group
  const int half = lane >> 4;     // 0/1
  const int l16  = lane & 15;

  const int n0  = blockIdx.z * 2;             // two images per block
  const int co0 = blockIdx.y * 64;
  const int bh  = blockIdx.x / 7;
  const int bw  = blockIdx.x % 7;
  const int h0  = bh * 8;
  const int w0  = bw * 8;

  const int img = wrow >> 1;                  // which staged patch
  // A-fragment source pixel rows within the 8x8 block (before kh/kw shift):
  // msub0 rows -> ph0, msub1 rows -> ph0+2
  const int ph0 = (wrow & 1) * 4 + (l16 >> 3);  // 0..5
  const int pw  = l16 & 7;                      // 0..7
  const int pbase = img * PIMG;

  v8f acc00 = {0.f,0.f,0.f,0.f,0.f,0.f,0.f,0.f};
  v8f acc01 = {0.f,0.f,0.f,0.f,0.f,0.f,0.f,0.f};
  v8f acc10 = {0.f,0.f,0.f,0.f,0.f,0.f,0.f,0.f};
  v8f acc11 = {0.f,0.f,0.f,0.f,0.f,0.f,0.f,0.f};

  for (int cb = 0; cb < C_IN; cb += 32) {
    __syncthreads();  // protect LDS from previous iteration's readers

    // ---- stage two input halo patches (10x10 spatial, 32 ch), f32 -> bf16
    for (int idx = tid; idx < 6400; idx += 256) {
      int im  = idx / 3200;
      int r2  = idx - im * 3200;
      int c   = r2 / 100;
      int rem = r2 - c * 100;
      int row = rem / 10;
      int col = rem - row * 10;
      int hh  = h0 - 1 + row;
      int ww  = w0 - 1 + col;
      float v = 0.f;
      if ((unsigned)hh < (unsigned)H_IN && (unsigned)ww < (unsigned)W_IN)
        v = x[(((n0 + im) * C_IN + cb + c) * H_IN + hh) * W_IN + ww];
      patch[im * PIMG + (row * 10 + col) * PCELL + c] = f2bf(v);
    }

    // ---- stage all 9 weight tiles, transposed to [co][c]
    for (int idx = tid; idx < KPOS * 32 * 8; idx += 256) {
      int pos = idx >> 8;          // 0..8
      int rem = idx & 255;
      int c   = rem >> 3;          // 0..31
      int co  = (rem & 7) * 8;     // 0,8,...,56
      union { uint4 q; unsigned short s[8]; } w;
      w.q = *reinterpret_cast<const uint4*>(
          &Wt[(pos * C_IN + cb + c) * C_OUT + co0 + co]);
#pragma unroll
      for (int j = 0; j < 8; ++j)
        btile[pos * BPOS + (co + j) * BROW + c] = w.s[j];
    }

    __syncthreads();

    // ---- 9 shifted WMMA steps: 4 WMMAs per tap from 8 ds_load_b128
#pragma unroll
    for (int pos = 0; pos < KPOS; ++pos) {
      const int kh = pos / 3;
      const int kw = pos % 3;

      // A 16x32 bf16 fragments for the two M-subtiles
      union { uint4 q[2]; v16bf v; } af0, af1;
      {
        const unsigned short* ap0 =
            &patch[pbase + ((ph0 + kh) * 10 + (pw + kw)) * PCELL + half * 8];
        af0.q[0] = *reinterpret_cast<const uint4*>(ap0);
        af0.q[1] = *reinterpret_cast<const uint4*>(ap0 + 16);
        const unsigned short* ap1 = ap0 + 2 * 10 * PCELL;  // ph0 + 2
        af1.q[0] = *reinterpret_cast<const uint4*>(ap1);
        af1.q[1] = *reinterpret_cast<const uint4*>(ap1 + 16);
      }

      // B 32x16 bf16 fragments for the two N-subtiles
      union { uint4 q[2]; v16bf v; } bf0, bf1;
      {
        const int col0 = wcol * 32 + l16;
        const unsigned short* bp0 =
            &btile[pos * BPOS + col0 * BROW + half * 8];
        bf0.q[0] = *reinterpret_cast<const uint4*>(bp0);
        bf0.q[1] = *reinterpret_cast<const uint4*>(bp0 + 16);
        const unsigned short* bp1 = bp0 + 16 * BROW;       // col0 + 16
        bf1.q[0] = *reinterpret_cast<const uint4*>(bp1);
        bf1.q[1] = *reinterpret_cast<const uint4*>(bp1 + 16);
      }

      acc00 = __builtin_amdgcn_wmma_f32_16x16x32_bf16(
          false, af0.v, false, bf0.v, (short)0, acc00, false, false);
      acc10 = __builtin_amdgcn_wmma_f32_16x16x32_bf16(
          false, af1.v, false, bf0.v, (short)0, acc10, false, false);
      acc01 = __builtin_amdgcn_wmma_f32_16x16x32_bf16(
          false, af0.v, false, bf1.v, (short)0, acc01, false, false);
      acc11 = __builtin_amdgcn_wmma_f32_16x16x32_bf16(
          false, af1.v, false, bf1.v, (short)0, acc11, false, false);
    }
  }

  // ---- epilogue: D layout lane=col, VGPR v -> row M = v + 8*half
  const int n  = n0 + img;
  const int hb = h0 + (wrow & 1) * 4 + half;   // msub0 row; msub1 = +2
#pragma unroll
  for (int msub = 0; msub < 2; ++msub) {
    const int h = hb + msub * 2;
#pragma unroll
    for (int nsub = 0; nsub < 2; ++nsub) {
      const int co = co0 + wcol * 32 + nsub * 16 + l16;
      const float bv = bias[co];
      v8f a = msub ? (nsub ? acc11 : acc10) : (nsub ? acc01 : acc00);
      float* op = &out[((n * C_OUT + co) * H_IN + h) * W_IN + w0];
      float4 s0 = make_float4(a[0] + bv, a[1] + bv, a[2] + bv, a[3] + bv);
      float4 s1 = make_float4(a[4] + bv, a[5] + bv, a[6] + bv, a[7] + bv);
      *reinterpret_cast<float4*>(op)     = s0;
      *reinterpret_cast<float4*>(op + 4) = s1;
    }
  }
}

extern "C" void kernel_launch(void* const* d_in, const int* in_sizes, int n_in,
                              void* d_out, int out_size, void* d_ws, size_t ws_size,
                              hipStream_t stream) {
  const float* x = (const float*)d_in[0];
  const float* W = (const float*)d_in[1];
  const float* b = (const float*)d_in[2];
  float* out = (float*)d_out;
  unsigned short* Wt = (unsigned short*)d_ws;  // 9*192*384*2 = 1.33 MB

  const int wt_elems = KPOS * C_IN * C_OUT;
  weight_transform<<<(wt_elems + 255) / 256, 256, 0, stream>>>(W, Wt);

  dim3 grid(7 * 7, C_OUT / 64, N_IMG / 2);  // (spatial, cout blocks, image pairs)
  conv3x3_wmma_bf16<<<grid, 256, 0, stream>>>(x, Wt, b, out);
}
